// WireDiscriminator_31696858644694
// MI455X (gfx1250) — compile-verified, matrix operand order checked by reading, not run on hardware
//
#include <hip/hip_runtime.h>
#include <hip/hip_bf16.h>

// ---------------------------------------------------------------------------
// Types for CDNA5 WMMA (gfx1250, wave32)
// ---------------------------------------------------------------------------
typedef __attribute__((ext_vector_type(16))) __bf16 v16bf;
typedef __attribute__((ext_vector_type(8)))  float  v8f;

union ABf {
    v16bf        v;
    unsigned int u[8];
};

__device__ inline unsigned short f2bf(float f) {
    unsigned int u = __builtin_bit_cast(unsigned int, f);
    unsigned int r = u + 0x7FFFu + ((u >> 16) & 1u);   // round-to-nearest-even
    return (unsigned short)(r >> 16);
}

__device__ inline float clamp01(float z) { return fminf(fmaxf(z, 0.f), 1.f); }
__device__ inline float hatk(float z)    { return fmaxf(0.f, 1.f - fabsf(z)); }

#define WMMA_BF16(A, Bf, C) \
    __builtin_amdgcn_wmma_f32_16x16x32_bf16(false, (A), false, (Bf), (short)0, (C), false, false)

// ---------------------------------------------------------------------------
// Constants
// ---------------------------------------------------------------------------
#define BATCH 64
#define NPTS  4096
#define CLSN  64
#define SPAT  100      // 10x10
#define FEATN 19200    // 192*100

// ---------------------------------------------------------------------------
// f32 -> bf16 weight conversion
// ---------------------------------------------------------------------------
__global__ void to_bf16(const float* __restrict__ src, unsigned short* __restrict__ dst, int n) {
    int i = blockIdx.x * 256 + threadIdx.x;
    if (i < n) dst[i] = f2bf(src[i]);
}

// ---------------------------------------------------------------------------
// Render + einsum:  R[b][c][s] = sum_n p[b][n][c] * val(theta[b][n], s)
// One block (256 thr / 8 waves) per batch. K blocked by 64 through LDS.
// GEMM view: M = class (64 -> 4 tiles), N = s padded to 128 (8 tiles),
// K = n (4096).  Wave w owns N-tile w; 4 M-tile accumulators per wave.
// Completely branch-free WMMA inner loop (EXEC all-ones everywhere).
// ---------------------------------------------------------------------------
__global__ __launch_bounds__(256) void render_gemm(const float* __restrict__ x,
                                                   const int* __restrict__ bern,
                                                   int use_mask,
                                                   float* __restrict__ R) {
    __shared__ unsigned short aLDS[64 * 64];    // [class][k]  (bf16 bits)
    __shared__ unsigned short vLDS[128 * 64];   // [s][k]      (bf16 bits), s padded

    const int b    = blockIdx.x;
    const int tid  = threadIdx.x;
    const int lane = tid & 31;
    const int wave = tid >> 5;

    // Zero the padded rows s = 100..127 once (never overwritten afterwards).
    for (int i = tid; i < 28 * 64; i += 256) vLDS[100 * 64 + i] = 0;

    v8f acc[4] = {};   // M-tiles 0..3 for this wave's N-tile

    const float* xb = x + (size_t)b * NPTS * 68;
    const int*   mb = bern + (size_t)b * NPTS;

    const int khalf = (lane >> 4) << 3;   // 0 or 8
    const int mrow  = lane & 15;

    for (int k0 = 0; k0 < NPTS; k0 += 64) {
        __syncthreads();   // prior WMMAs done before LDS overwrite

        // ---- stage A tile: aLDS[c][k] = p[b][k0+k][c] * mask ----
        {
            int nloc = tid >> 2;                 // 0..63
            int n    = k0 + nloc;
            int cb   = (tid & 3) * 16;
            const float* row = xb + (size_t)n * 68;
            float msk = use_mask ? (float)mb[n] : 1.0f;
#pragma unroll
            for (int i = 0; i < 16; ++i) {
                int c = cb + i;
                aLDS[c * 64 + nloc] = f2bf(row[c] * msk);
            }
        }
        // ---- stage B tile: vLDS[s][k] = val(theta[k0+k], s) ----
        {
            int nloc = tid & 63;
            int n    = k0 + nloc;
            const float* th = xb + (size_t)n * 68 + 64;
            float t0 = th[0] * 10.f, t1 = th[1] * 10.f;
            float t2 = th[2] * 10.f, t3 = th[3] * 10.f;
            for (int s = (tid >> 6); s < SPAT; s += 4) {
                float gx = (float)(s / 10), gy = (float)(s % 10);
                float by = clamp01(gy - t1) * clamp01(t3 - gy);
                float bx = clamp01(gx - t0) * clamp01(t2 - gx);
                float F0 = hatk(gx - t0) * by;
                float F1 = hatk(gx - t2) * by;
                float F2 = hatk(gy - t1) * bx;
                float F3 = hatk(gy - t3) * bx;
                vLDS[s * 64 + nloc] = f2bf(fmaxf(fmaxf(F0, F1), fmaxf(F2, F3)));
            }
        }
        __syncthreads();

        // ---- WMMA over the staged 64-wide K block (2 sub-steps of 32) ----
#pragma unroll
        for (int ks = 0; ks < 64; ks += 32) {
            ABf bf;   // B fragment for this wave's N-tile (shared by all M-tiles)
#pragma unroll
            for (int j = 0; j < 8; ++j) {
                int kk = ks + 2 * j + ((j >= 4) ? 8 : 0) + khalf;
                bf.u[j] = *(const unsigned int*)&vLDS[(wave * 16 + mrow) * 64 + kk];
            }
#pragma unroll
            for (int mt = 0; mt < 4; ++mt) {
                ABf a;
#pragma unroll
                for (int j = 0; j < 8; ++j) {
                    int kk = ks + 2 * j + ((j >= 4) ? 8 : 0) + khalf;
                    a.u[j] = *(const unsigned int*)&aLDS[(mt * 16 + mrow) * 64 + kk];
                }
                acc[mt] = WMMA_BF16(a.v, bf.v, acc[mt]);
            }
        }
    }

    // ---- write D:  lanes 0-15 -> N=lane, M=r ; lanes 16-31 -> N=lane-16, M=r+8
    float* Rb   = R + (size_t)b * CLSN * SPAT;
    int    ncol = lane & 15;
    int    mofs = (lane >> 4) * 8;
    int    s    = wave * 16 + ncol;
    if (s < SPAT) {
#pragma unroll
        for (int mt = 0; mt < 4; ++mt) {
#pragma unroll
            for (int r = 0; r < 8; ++r) {
                int c = mt * 16 + mofs + r;
                Rb[c * SPAT + s] = acc[mt][r];
            }
        }
    }
}

// ---------------------------------------------------------------------------
// 3x3 conv (pad 1) as implicit GEMM.  M = spatial padded to 128 (8 tiles),
// N = Cout tile of 16, K = Cin*9.  Grid (B, Cout/16), 128 threads (4 waves),
// image in LDS.  Wave w owns M-tiles w and w+4 (tile 7 is all-zero padding).
// Output includes bias (pre-BN, matching reference).
// ---------------------------------------------------------------------------
__global__ __launch_bounds__(128) void conv_wmma(const float* __restrict__ in,        // [B][Cin][100]
                                                 const unsigned short* __restrict__ wbf, // [Cout][Cin*9] bf16
                                                 const float* __restrict__ bias,
                                                 float* __restrict__ out,             // [B][Cout][100]
                                                 int Cin, int Cout) {
    extern __shared__ unsigned short img[];   // [Cin][12][12] zero-padded bf16

    const int b    = blockIdx.x;
    const int co0  = blockIdx.y * 16;
    const int tid  = threadIdx.x;
    const int lane = tid & 31;
    const int wave = tid >> 5;

    const float* inb = in + (size_t)b * Cin * SPAT;
    const int total = Cin * 144;
    for (int i = tid; i < total; i += 128) {
        int ci = i / 144, r = i % 144;
        int X = r / 12, Y = r % 12;
        float v = 0.f;
        if (X >= 1 && X <= 10 && Y >= 1 && Y <= 10)
            v = inb[ci * SPAT + (X - 1) * 10 + (Y - 1)];
        img[i] = f2bf(v);
    }
    __syncthreads();

    const int K     = Cin * 9;                // 576 or 1152, 32-divisible
    const int khalf = (lane >> 4) << 3;
    const int mrow  = lane & 15;

    v8f acc[2] = {};

    const unsigned short* wrow = wbf + (size_t)(co0 + mrow) * K;

    for (int kb = 0; kb < K; kb += 32) {
        ABf bf;
#pragma unroll
        for (int j = 0; j < 8; ++j) {
            int kk = kb + 2 * j + ((j >= 4) ? 8 : 0) + khalf;
            bf.u[j] = *(const unsigned int*)&wrow[kk];
        }
#pragma unroll
        for (int t = 0; t < 2; ++t) {
            int mt = wave + 4 * t;            // 0..7 (mt==7 -> zero tile)
            int s  = mt * 16 + mrow;
            ABf a;
            if (s < SPAT) {
                int u = s / 10, v = s % 10;
#pragma unroll
                for (int j = 0; j < 8; ++j) {
                    int kk  = kb + 2 * j + ((j >= 4) ? 8 : 0) + khalf;
                    int ci0 = kk / 9,        r0 = kk - ci0 * 9;
                    int ci1 = (kk + 1) / 9,  r1 = (kk + 1) - ci1 * 9;
                    unsigned int lo = img[(ci0 * 12 + (u + r0 / 3)) * 12 + (v + r0 % 3)];
                    unsigned int hi = img[(ci1 * 12 + (u + r1 / 3)) * 12 + (v + r1 % 3)];
                    a.u[j] = lo | (hi << 16);
                }
            } else {
#pragma unroll
                for (int j = 0; j < 8; ++j) a.u[j] = 0u;
            }
            acc[t] = WMMA_BF16(a.v, bf.v, acc[t]);   // no exec-masking around WMMA
        }
    }

    float* ob   = out + ((size_t)b * Cout + co0) * SPAT;
    int    ncol = lane & 15;
    int    mofs = (lane >> 4) * 8;
    float  bsv  = bias[co0 + ncol];
#pragma unroll
    for (int t = 0; t < 2; ++t) {
        int mt = wave + 4 * t;
#pragma unroll
        for (int r = 0; r < 8; ++r) {
            int s = mt * 16 + mofs + r;
            if (s < SPAT) ob[ncol * SPAT + s] = acc[t][r] + bsv;
        }
    }
}

// ---------------------------------------------------------------------------
// BatchNorm statistics per channel over (B, spatial) = 6400 elements.
// ---------------------------------------------------------------------------
__global__ __launch_bounds__(256) void bn_stats(const float* __restrict__ y,
                                                float* __restrict__ mu,
                                                float* __restrict__ rs, int Cout) {
    __shared__ float s1[256], s2[256];
    int c = blockIdx.x, tid = threadIdx.x;
    float a = 0.f, q = 0.f;
    for (int i = tid; i < BATCH * SPAT; i += 256) {
        int bb = i / SPAT, s = i % SPAT;
        float v = y[((size_t)bb * Cout + c) * SPAT + s];
        a += v; q += v * v;
    }
    s1[tid] = a; s2[tid] = q;
    __syncthreads();
    for (int st = 128; st > 0; st >>= 1) {
        if (tid < st) { s1[tid] += s1[tid + st]; s2[tid] += s2[tid + st]; }
        __syncthreads();
    }
    if (tid == 0) {
        float m = s1[0] / 6400.f;
        float v = s2[0] / 6400.f - m * m;
        mu[c] = m;
        rs[c] = rsqrtf(v + 1e-5f);
    }
}

__global__ void bn_apply(float* __restrict__ y, const float* __restrict__ mu,
                         const float* __restrict__ rs, const float* __restrict__ g,
                         const float* __restrict__ be, int Cout, int total) {
    int i = blockIdx.x * 256 + threadIdx.x;
    if (i >= total) return;
    int c = (i / SPAT) % Cout;
    float v = (y[i] - mu[c]) * rs[c] * g[c] + be[c];
    y[i] = v > 0.f ? v : 0.f;
}

// ---------------------------------------------------------------------------
// Head GEMM:  H[64][128] = relu(feat[64][19200] @ W^T + b),  W bf16 [128][19200]
// One wave per block, one N-tile per block (grid 8), 4 M-tile accumulators.
// ---------------------------------------------------------------------------
__global__ __launch_bounds__(32) void head_gemm(const float* __restrict__ feat,
                                                const unsigned short* __restrict__ wbf,
                                                const float* __restrict__ bias,
                                                float* __restrict__ H) {
    const int nt    = blockIdx.x;             // 0..7
    const int lane  = threadIdx.x & 31;
    const int khalf = (lane >> 4) << 3;
    const int l15   = lane & 15;

    v8f acc[4] = {};
    const unsigned short* wrow = wbf + (size_t)(nt * 16 + l15) * FEATN;

    for (int kb = 0; kb < FEATN; kb += 32) {
        // stream-prefetch the weight row ahead (global_prefetch_b8)
        __builtin_prefetch(wrow + kb + 1024, 0, 0);
        ABf bf;
#pragma unroll
        for (int j = 0; j < 8; ++j) {
            int kk = kb + 2 * j + ((j >= 4) ? 8 : 0) + khalf;
            bf.u[j] = *(const unsigned int*)&wrow[kk];
        }
#pragma unroll
        for (int t = 0; t < 4; ++t) {
            const float* frow = feat + (size_t)(t * 16 + l15) * FEATN;
            ABf a;
#pragma unroll
            for (int j = 0; j < 8; ++j) {
                int kk = kb + 2 * j + ((j >= 4) ? 8 : 0) + khalf;
                unsigned int lo = f2bf(frow[kk]);
                unsigned int hi = f2bf(frow[kk + 1]);
                a.u[j] = lo | (hi << 16);
            }
            acc[t] = WMMA_BF16(a.v, bf.v, acc[t]);
        }
    }

    int mofs = (lane >> 4) * 8;
#pragma unroll
    for (int t = 0; t < 4; ++t) {
#pragma unroll
        for (int r = 0; r < 8; ++r) {
            int m = t * 16 + mofs + r;        // batch row
            int n = nt * 16 + l15;            // hidden unit
            float v = acc[t][r] + bias[n];
            H[m * 128 + n] = v > 0.f ? v : 0.f;
        }
    }
}

// ---------------------------------------------------------------------------
// Small heads
// ---------------------------------------------------------------------------
__global__ void pred_head(const float* __restrict__ H, const float* __restrict__ w2,
                          const float* __restrict__ b2, float* __restrict__ out) {
    int b = threadIdx.x;
    if (b < BATCH) {
        float s = b2[0];
        for (int j = 0; j < 128; ++j) s += H[b * 128 + j] * w2[j];
        out[b] = 1.f / (1.f + expf(-s));
    }
}

__global__ void score_head(const float* __restrict__ H, const float* __restrict__ w2,
                           const float* __restrict__ b2, float* __restrict__ out) {
    int i = blockIdx.x * 256 + threadIdx.x;
    if (i >= BATCH * CLSN) return;
    int b = i >> 6, c = i & 63;
    float s = b2[c];
    for (int j = 0; j < 128; ++j) s += H[b * 128 + j] * w2[c * 128 + j];
    out[i] = s;
}

// ---------------------------------------------------------------------------
// Orchestration
// ---------------------------------------------------------------------------
extern "C" void kernel_launch(void* const* d_in, const int* in_sizes, int n_in,
                              void* d_out, int out_size, void* d_ws, size_t ws_size,
                              hipStream_t stream) {
    (void)in_sizes; (void)n_in; (void)out_size; (void)ws_size;

    const float* x    = (const float*)d_in[0];
    const int*   bern = (const int*)d_in[1];
    // gconv at 2..13, lconv at 14..25 : order w1,b1,g1,be1,w2,b2,g2,be2,w3,b3,g3,be3
    const float* fc_w1  = (const float*)d_in[26];
    const float* fc_b1  = (const float*)d_in[27];
    const float* fc_w2  = (const float*)d_in[28];
    const float* fc_b2  = (const float*)d_in[29];
    const float* dec_w1 = (const float*)d_in[30];
    const float* dec_b1 = (const float*)d_in[31];
    const float* dec_w2 = (const float*)d_in[32];
    const float* dec_b2 = (const float*)d_in[33];

    float* out = (float*)d_out;

    // --- workspace layout (bytes, 256-aligned) ---
    char*  ws  = (char*)d_ws;
    size_t off = 0;
    auto carve = [&](size_t bytes) -> void* {
        void* p = ws + off;
        off += (bytes + 255) & ~(size_t)255;
        return p;
    };
    const int NW1 = 64 * 64 * 9;      // 36864
    const int NW2 = 128 * 64 * 9;     // 73728
    const int NW3 = 192 * 128 * 9;    // 221184
    const int NWF = 128 * FEATN;      // 2457600

    unsigned short* wgbf[3] = { (unsigned short*)carve(NW1 * 2),
                                (unsigned short*)carve(NW2 * 2),
                                (unsigned short*)carve(NW3 * 2) };
    unsigned short* wlbf[3] = { (unsigned short*)carve(NW1 * 2),
                                (unsigned short*)carve(NW2 * 2),
                                (unsigned short*)carve(NW3 * 2) };
    unsigned short* wfcbf  = (unsigned short*)carve((size_t)NWF * 2);
    unsigned short* wdecbf = (unsigned short*)carve((size_t)NWF * 2);

    float* R  = (float*)carve((size_t)BATCH * 64  * SPAT * 4);
    float* y1 = (float*)carve((size_t)BATCH * 64  * SPAT * 4);
    float* y2 = (float*)carve((size_t)BATCH * 128 * SPAT * 4);
    float* y3 = (float*)carve((size_t)BATCH * 192 * SPAT * 4);   // == feat
    float* mu = (float*)carve(192 * 4);
    float* rs = (float*)carve(192 * 4);
    float* H  = (float*)carve(BATCH * 128 * 4);

    // --- weight conversion (recomputed each call; deterministic) ---
    auto cvt = [&](const float* src, unsigned short* dst, int n) {
        to_bf16<<<(n + 255) / 256, 256, 0, stream>>>(src, dst, n);
    };
    cvt((const float*)d_in[2],  wgbf[0], NW1);
    cvt((const float*)d_in[6],  wgbf[1], NW2);
    cvt((const float*)d_in[10], wgbf[2], NW3);
    cvt((const float*)d_in[14], wlbf[0], NW1);
    cvt((const float*)d_in[18], wlbf[1], NW2);
    cvt((const float*)d_in[22], wlbf[2], NW3);
    cvt(fc_w1,  wfcbf,  NWF);
    cvt(dec_w1, wdecbf, NWF);

    const int Cin[3]  = { 64, 64, 128 };
    const int Cout[3] = { 64, 128, 192 };
    float* ybuf[3] = { y1, y2, y3 };

    for (int pass = 0; pass < 2; ++pass) {
        unsigned short** wbf = pass ? wlbf : wgbf;
        int pbase = 2 + pass * 12;

        // render (fused val rasterization + WMMA einsum)
        render_gemm<<<BATCH, 256, 0, stream>>>(x, bern, pass, R);

        // conv block
        const float* cur = R;
        for (int l = 0; l < 3; ++l) {
            const float* bia = (const float*)d_in[pbase + l * 4 + 1];
            const float* gam = (const float*)d_in[pbase + l * 4 + 2];
            const float* bet = (const float*)d_in[pbase + l * 4 + 3];
            int ci = Cin[l], co = Cout[l];
            int tot = BATCH * co * SPAT;
            conv_wmma<<<dim3(BATCH, co / 16), 128, (size_t)ci * 144 * 2, stream>>>(
                cur, wbf[l], bia, ybuf[l], ci, co);
            bn_stats<<<co, 256, 0, stream>>>(ybuf[l], mu, rs, co);
            bn_apply<<<(tot + 255) / 256, 256, 0, stream>>>(ybuf[l], mu, rs, gam, bet, co, tot);
            cur = ybuf[l];
        }

        float* pred_out  = out + (size_t)pass * (BATCH + BATCH * CLSN);
        float* score_out = pred_out + BATCH;

        // heads: feat = y3 (layout [b][c][s] == reference reshape)
        head_gemm<<<8, 32, 0, stream>>>(y3, wfcbf, fc_b1, H);
        pred_head<<<1, 64, 0, stream>>>(H, fc_w2, fc_b2, pred_out);

        head_gemm<<<8, 32, 0, stream>>>(y3, wdecbf, dec_b1, H);
        score_head<<<(BATCH * CLSN + 255) / 256, 256, 0, stream>>>(H, dec_w2, dec_b2, score_out);
    }
}
